// PortfolioGRU_32427003085210
// MI455X (gfx1250) — matrix operand order repaired and checked by reading, not compiled, not run on hardware
//
#include <hip/hip_runtime.h>
#include <math.h>

// ---------------- types ----------------
typedef __bf16 bf16_t;
typedef __attribute__((ext_vector_type(16))) __bf16 v16bf;
typedef __attribute__((ext_vector_type(8)))  __bf16 v8bf;
typedef __attribute__((ext_vector_type(8)))  float  v8f;

#define WMMA_BF16(a, b, c) \
    __builtin_amdgcn_wmma_f32_16x16x32_bf16(false, (a), false, (b), (short)0, (c), false, false)

#define BB   512
#define TT   128
#define NN   500
#define HH   1024
#define NPAD 512   // 500 padded to 512

// ---------------- WMMA fragment loaders ----------------
// A fragment (16x32 bf16, M x K), row-major source with leading dim lda.
// lane L: row = m0 + (L&15); elements 0..7 <- K = k + s .. k+s+7, elements
// 8..15 <- K = k + 16 + s .. , where s = (L>=16)*8.  Two 16B loads.
static __device__ inline v16bf ld_frag_a(const bf16_t* __restrict__ base,
                                         int m0, long lda, int k) {
    int lane = threadIdx.x & 31;
    int m = m0 + (lane & 15);
    int s = (lane >> 4) << 3;
    const bf16_t* p = base + (long)m * lda + k + s;
    v8bf lo = *(const v8bf*)p;
    v8bf hi = *(const v8bf*)(p + 16);
    v16bf f;
#pragma unroll
    for (int i = 0; i < 8; ++i) { f[i] = lo[i]; f[i + 8] = hi[i]; }
    return f;
}

// B fragment (32x16 bf16, K x N) where B[k,n] = W[n*ldw + k] (W row-major NxK).
// lane L: col n = n0 + (L&15); elements 0..7 <- K = k + ks .. k+ks+7,
// elements 8..15 <- K = k + 16 + ks .., ks = (L>=16)*8.  Two 16B loads.
static __device__ inline v16bf ld_frag_b(const bf16_t* __restrict__ W,
                                         int n0, long ldw, int k) {
    int lane = threadIdx.x & 31;
    int n = n0 + (lane & 15);
    int ks = (lane >> 4) << 3;
    const bf16_t* p = W + (long)n * ldw + k + ks;
    v8bf lo = *(const v8bf*)p;
    v8bf hi = *(const v8bf*)(p + 16);
    v16bf f;
#pragma unroll
    for (int i = 0; i < 8; ++i) { f[i] = lo[i]; f[i + 8] = hi[i]; }
    return f;
}

static __device__ inline float sigmoidf_(float x) {
    return 1.0f / (1.0f + __expf(-x));
}

// ---------------- f32 -> bf16 convert with zero padding ----------------
__global__ __launch_bounds__(256) void cvt_pad_kernel(
    const float* __restrict__ src, int R, int C,
    bf16_t* __restrict__ dst, int Rp, int Cp) {
    size_t idx = (size_t)blockIdx.x * 256 + threadIdx.x;
    size_t total = (size_t)Rp * Cp;
    if (idx >= total) return;
    int r = (int)(idx / Cp);
    int c = (int)(idx % Cp);
    float v = (r < R && c < C) ? src[(size_t)r * C + c] : 0.0f;
    dst[idx] = (bf16_t)v;
}

// ---------------- fused GRU step ----------------
// Per wave: one 32x16 (batch x hidden) tile = 2 M sub-tiles sharing the same
// three B (weight) fragments -> 6 WMMAs per 10 b128 loads per K-step.
// Accumulators per sub-tile:
//   acc_r : x@Wih_r + h@Whh_r      acc_z : x@Wih_z + h@Whh_z
//   acc_nx: x@Wih_n                acc_nh: h@Whh_n  (scaled by r later)
__global__ __launch_bounds__(128) void gru_step_kernel(
    const bf16_t* __restrict__ xbase, long xstride, int Kx,
    const bf16_t* __restrict__ wih, long ldwih,
    const bf16_t* __restrict__ whh,
    const bf16_t* __restrict__ hprev_bf,
    const float*  __restrict__ hprev_f,
    const float*  __restrict__ b_ih, const float* __restrict__ b_hh,
    float* __restrict__ hnext_f, bf16_t* __restrict__ hnext_bf,
    bf16_t* __restrict__ seq_out) {
    const int H = HH;
    int wave = threadIdx.x >> 5;
    int lane = threadIdx.x & 31;
    int n0 = (blockIdx.x * 4 + wave) * 16;   // hidden tile [0,1024)
    int m0 = blockIdx.y * 32;                // batch tile  [0,512), 2 sub-tiles

    v8f acc_r[2] = {}, acc_z[2] = {}, acc_nx[2] = {}, acc_nh[2] = {};

    for (int k = 0; k < Kx; k += 32) {
        v16bf a0 = ld_frag_a(xbase, m0,      xstride, k);
        v16bf a1 = ld_frag_a(xbase, m0 + 16, xstride, k);
        v16bf br = ld_frag_b(wih, n0,         ldwih, k);
        v16bf bz = ld_frag_b(wih, n0 + H,     ldwih, k);
        v16bf bn = ld_frag_b(wih, n0 + 2 * H, ldwih, k);
        acc_r[0]  = WMMA_BF16(a0, br, acc_r[0]);
        acc_r[1]  = WMMA_BF16(a1, br, acc_r[1]);
        acc_z[0]  = WMMA_BF16(a0, bz, acc_z[0]);
        acc_z[1]  = WMMA_BF16(a1, bz, acc_z[1]);
        acc_nx[0] = WMMA_BF16(a0, bn, acc_nx[0]);
        acc_nx[1] = WMMA_BF16(a1, bn, acc_nx[1]);
    }
    for (int k = 0; k < H; k += 32) {
        v16bf a0 = ld_frag_a(hprev_bf, m0,      H, k);
        v16bf a1 = ld_frag_a(hprev_bf, m0 + 16, H, k);
        v16bf br = ld_frag_b(whh, n0,         H, k);
        v16bf bz = ld_frag_b(whh, n0 + H,     H, k);
        v16bf bn = ld_frag_b(whh, n0 + 2 * H, H, k);
        acc_r[0]  = WMMA_BF16(a0, br, acc_r[0]);
        acc_r[1]  = WMMA_BF16(a1, br, acc_r[1]);
        acc_z[0]  = WMMA_BF16(a0, bz, acc_z[0]);
        acc_z[1]  = WMMA_BF16(a1, bz, acc_z[1]);
        acc_nh[0] = WMMA_BF16(a0, bn, acc_nh[0]);
        acc_nh[1] = WMMA_BF16(a1, bn, acc_nh[1]);
    }

    // C/D layout: lane L, VGPR i -> row m0 + i + (L>=16)*8, col n0 + (L&15)
    int j = n0 + (lane & 15);
    float bir = b_ih[j],         bhr = b_hh[j];
    float biz = b_ih[j + H],     bhz = b_hh[j + H];
    float bin = b_ih[j + 2 * H], bhn = b_hh[j + 2 * H];
#pragma unroll
    for (int sub = 0; sub < 2; ++sub) {
        int mr0 = m0 + sub * 16 + ((lane >> 4) << 3);
#pragma unroll
        for (int i = 0; i < 8; ++i) {
            int m = mr0 + i;
            float r  = sigmoidf_(acc_r[sub][i] + bir + bhr);
            float zg = sigmoidf_(acc_z[sub][i] + biz + bhz);
            float n  = tanhf(acc_nx[sub][i] + bin + r * (acc_nh[sub][i] + bhn));
            float ho = hprev_f[(size_t)m * H + j];
            float hn = (1.0f - zg) * n + zg * ho;
            hnext_f[(size_t)m * H + j]  = hn;
            bf16_t hbv = (bf16_t)hn;
            hnext_bf[(size_t)m * H + j] = hbv;
            if (seq_out) seq_out[(size_t)m * H + j] = hbv;
        }
    }
}

// ---------------- generic WMMA GEMM: C = A @ W^T + bias ----------------
// 32x16 output per wave (2 M sub-tiles share the B fragment).
__global__ __launch_bounds__(128) void gemm_bias_kernel(
    const bf16_t* __restrict__ A, long lda,
    const bf16_t* __restrict__ W, long ldw,
    const float* __restrict__ bias,
    float* __restrict__ C, int ldc, int K, int Ncap) {
    int wave = threadIdx.x >> 5;
    int lane = threadIdx.x & 31;
    int n0 = (blockIdx.x * 4 + wave) * 16;
    int m0 = blockIdx.y * 32;
    v8f acc[2] = {};
    for (int k = 0; k < K; k += 32) {
        v16bf a0 = ld_frag_a(A, m0,      lda, k);
        v16bf a1 = ld_frag_a(A, m0 + 16, lda, k);
        v16bf b  = ld_frag_b(W, n0, ldw, k);
        acc[0] = WMMA_BF16(a0, b, acc[0]);
        acc[1] = WMMA_BF16(a1, b, acc[1]);
    }
    int j = n0 + (lane & 15);
    if (j < Ncap) {
        float bj = bias[j];
#pragma unroll
        for (int sub = 0; sub < 2; ++sub) {
            int mr0 = m0 + sub * 16 + ((lane >> 4) << 3);
#pragma unroll
            for (int i = 0; i < 8; ++i)
                C[(size_t)(mr0 + i) * ldc + j] = acc[sub][i] + bj;
        }
    }
}

// ---------------- LayerNorm + SiLU, output bf16 ----------------
__global__ __launch_bounds__(256) void ln_silu_kernel(
    const float* __restrict__ z, const float* __restrict__ g,
    const float* __restrict__ bb, bf16_t* __restrict__ out) {
    int row = blockIdx.x, tid = threadIdx.x;
    __shared__ float s1[256], s2[256];
    float vals[4], sum = 0.f, sq = 0.f;
#pragma unroll
    for (int i = 0; i < 4; ++i) {
        int c = tid + i * 256;
        float v = z[(size_t)row * HH + c];
        vals[i] = v; sum += v; sq += v * v;
    }
    s1[tid] = sum; s2[tid] = sq; __syncthreads();
    for (int s = 128; s > 0; s >>= 1) {
        if (tid < s) { s1[tid] += s1[tid + s]; s2[tid] += s2[tid + s]; }
        __syncthreads();
    }
    float mu = s1[0] / (float)HH;
    float var = s2[0] / (float)HH - mu * mu;
    float rstd = rsqrtf(var + 1e-5f);
#pragma unroll
    for (int i = 0; i < 4; ++i) {
        int c = tid + i * 256;
        float y = (vals[i] - mu) * rstd * g[c] + bb[c];
        float s = y * sigmoidf_(y);
        out[(size_t)row * HH + c] = (bf16_t)s;
    }
}

// ---------------- masked softmax + 20-iter rebalance ----------------
__global__ __launch_bounds__(512) void softmax_rebalance_kernel(
    const float* __restrict__ scores, int lds, float* __restrict__ out) {
    int b = blockIdx.x;
    int tid = threadIdx.x;
    __shared__ float red[512];

    auto rsum = [&](float v) -> float {
        red[tid] = v; __syncthreads();
        for (int s = 256; s > 0; s >>= 1) {
            if (tid < s) red[tid] += red[tid + s];
            __syncthreads();
        }
        float r = red[0]; __syncthreads();
        return r;
    };
    auto rmax = [&](float v) -> float {
        red[tid] = v; __syncthreads();
        for (int s = 256; s > 0; s >>= 1) {
            if (tid < s) red[tid] = fmaxf(red[tid], red[tid + s]);
            __syncthreads();
        }
        float r = red[0]; __syncthreads();
        return r;
    };

    bool act = tid < NN;
    float s = act ? scores[(size_t)b * lds + tid] : -INFINITY;
    if (s < 0.0f) s = -INFINITY;                 // mask
    float mx = rmax(s);
    float e = (s == -INFINITY) ? 0.0f : __expf(s - mx);
    if (!act) e = 0.0f;
    float tot = rsum(e);
    float w = e / tot;                            // softmax
    w = fminf(fmaxf(w, 0.0f), 0.1f);              // clip(lb,ub)

    for (int it = 0; it < 20; ++it) {
        float excess = rsum(act ? w : 0.0f) - 1.0f;
        float surplus = fmaxf(w - 0.1f, 0.0f);
        float totsur = rsum(act ? surplus : 0.0f);
        float upd = (totsur > 0.0f)
                        ? w - surplus / fmaxf(totsur, 1e-12f) * excess
                        : w - excess / (float)NN;
        upd = fminf(fmaxf(upd, 0.0f), 0.1f);
        if (excess > 1e-6f) w = upd;              // row-uniform condition
    }
    if (act) out[(size_t)b * NN + tid] = w;
}

// ---------------- host launcher ----------------
extern "C" void kernel_launch(void* const* d_in, const int* in_sizes, int n_in,
                              void* d_out, int out_size, void* d_ws, size_t ws_size,
                              hipStream_t stream) {
    const float* x     = (const float*)d_in[0];
    const float* w_ih0 = (const float*)d_in[1];
    const float* w_hh0 = (const float*)d_in[2];
    const float* b_ih0 = (const float*)d_in[3];
    const float* b_hh0 = (const float*)d_in[4];
    const float* w_ih1 = (const float*)d_in[5];
    const float* w_hh1 = (const float*)d_in[6];
    const float* b_ih1 = (const float*)d_in[7];
    const float* b_hh1 = (const float*)d_in[8];
    const float* w1    = (const float*)d_in[9];
    const float* b1    = (const float*)d_in[10];
    const float* ln_g  = (const float*)d_in[11];
    const float* ln_b  = (const float*)d_in[12];
    const float* w2    = (const float*)d_in[13];
    const float* b2    = (const float*)d_in[14];

    // workspace carve-up (256B aligned)
    char* ws = (char*)d_ws;
    size_t off = 0;
    auto carve = [&](size_t bytes) -> char* {
        char* p = ws + off;
        off = (off + bytes + 255) & ~(size_t)255;
        return p;
    };
    bf16_t* xbf   = (bf16_t*)carve((size_t)BB * TT * NPAD * 2); // 64 MB
    bf16_t* wih0b = (bf16_t*)carve((size_t)3 * HH * NPAD * 2);
    bf16_t* whh0b = (bf16_t*)carve((size_t)3 * HH * HH * 2);
    bf16_t* wih1b = (bf16_t*)carve((size_t)3 * HH * HH * 2);
    bf16_t* whh1b = (bf16_t*)carve((size_t)3 * HH * HH * 2);
    bf16_t* w1b   = (bf16_t*)carve((size_t)HH * HH * 2);
    bf16_t* w2b   = (bf16_t*)carve((size_t)NPAD * HH * 2);      // rows 500..511 = 0
    bf16_t* hs0   = (bf16_t*)carve((size_t)TT * BB * HH * 2);   // 134 MB, [T][B][H]
    float*  hf[2]; bf16_t* hb[2];
    hf[0] = (float*)carve((size_t)BB * HH * 4);
    hf[1] = (float*)carve((size_t)BB * HH * 4);
    hb[0] = (bf16_t*)carve((size_t)BB * HH * 2);
    hb[1] = (bf16_t*)carve((size_t)BB * HH * 2);
    float*  zbuf   = (float*)carve((size_t)BB * HH * 4);
    bf16_t* sbf    = (bf16_t*)carve((size_t)BB * HH * 2);
    float*  scores = (float*)carve((size_t)BB * 512 * 4);

    auto cvt = [&](const float* src, int R, int C, bf16_t* dst, int Rp, int Cp) {
        size_t total = (size_t)Rp * Cp;
        int grid = (int)((total + 255) / 256);
        cvt_pad_kernel<<<grid, 256, 0, stream>>>(src, R, C, dst, Rp, Cp);
    };

    // bf16 conversions (weights once; all stay resident in L2 thereafter)
    cvt(x,     BB * TT, NN, xbf,   BB * TT, NPAD);
    cvt(w_ih0, 3 * HH,  NN, wih0b, 3 * HH,  NPAD);
    cvt(w_hh0, 3 * HH,  HH, whh0b, 3 * HH,  HH);
    cvt(w_ih1, 3 * HH,  HH, wih1b, 3 * HH,  HH);
    cvt(w_hh1, 3 * HH,  HH, whh1b, 3 * HH,  HH);
    cvt(w1,    HH,      HH, w1b,   HH,      HH);
    cvt(w2,    NN,      HH, w2b,   NPAD,    HH);

    dim3 sgrid(16, 16);   // 64 hidden tiles / 4 waves, 16 batch tiles (32 rows each)
    dim3 sblk(128);

    // ---- layer 0 ----
    hipMemsetAsync(hf[0], 0, (size_t)BB * HH * 4, stream);
    hipMemsetAsync(hb[0], 0, (size_t)BB * HH * 2, stream);
    for (int t = 0; t < TT; ++t) {
        int cur = t & 1, nxt = cur ^ 1;
        gru_step_kernel<<<sgrid, sblk, 0, stream>>>(
            xbf + (size_t)t * NPAD, (long)TT * NPAD, NPAD,
            wih0b, (long)NPAD, whh0b,
            hb[cur], hf[cur], b_ih0, b_hh0,
            hf[nxt], hb[nxt], hs0 + (size_t)t * BB * HH);
    }

    // ---- layer 1 ----
    hipMemsetAsync(hf[0], 0, (size_t)BB * HH * 4, stream);
    hipMemsetAsync(hb[0], 0, (size_t)BB * HH * 2, stream);
    for (int t = 0; t < TT; ++t) {
        int cur = t & 1, nxt = cur ^ 1;
        gru_step_kernel<<<sgrid, sblk, 0, stream>>>(
            hs0 + (size_t)t * BB * HH, (long)HH, HH,
            wih1b, (long)HH, whh1b,
            hb[cur], hf[cur], b_ih1, b_hh1,
            hf[nxt], hb[nxt], (bf16_t*)nullptr);
    }
    // T even -> final h in buffers index 0 (hb[0], hf[0])

    // ---- head ----
    gemm_bias_kernel<<<dim3(16, 16), 128, 0, stream>>>(
        hb[0], (long)HH, w1b, (long)HH, b1, zbuf, HH, HH, HH);
    ln_silu_kernel<<<BB, 256, 0, stream>>>(zbuf, ln_g, ln_b, sbf);
    gemm_bias_kernel<<<dim3(8, 16), 128, 0, stream>>>(
        sbf, (long)HH, w2b, (long)HH, b2, scores, 512, HH, NN);
    softmax_rebalance_kernel<<<BB, 512, 0, stream>>>(scores, 512, (float*)d_out);
}
